// MultiHeadSelfAttn_37211596653230
// MI455X (gfx1250) — matrix operand order repaired and checked by reading, not compiled, not run on hardware
//
#include <hip/hip_runtime.h>

#define BB   2
#define SS   2048
#define DDIM 1024
#define HH   16
#define HDIM 64
#define SM_SCALE 0.125f   // 1/sqrt(64), folded into Q projection

typedef __attribute__((ext_vector_type(16))) __bf16 v16bf;
typedef __attribute__((ext_vector_type(8)))  float  v8f;

union FragU { v16bf v; uint4 q[2]; };

// --- helpers -------------------------------------------------------------

__device__ inline unsigned short f2bf(float x) {
    return __builtin_bit_cast(unsigned short, (__bf16)x);  // v_cvt, RNE
}

// A-fragment (16x32 bf16) from row-major [M x ld]: lane l holds row m=l%16,
// k = 16g+e -> two 16B chunks at +16g and +16g+8.
__device__ inline v16bf load_frag_A(const unsigned short* base, int ld, int k0, int lane) {
    const int m = lane & 15, g = lane >> 4;
    const unsigned short* p = base + (size_t)m * ld + k0 + 16 * g;
    FragU f;
    f.q[0] = *(const uint4*)(p);
    f.q[1] = *(const uint4*)(p + 8);
    return f.v;
}

// B-fragment (32x16 bf16) where k runs contiguously in memory (K^T from
// row-major K, transposed weights, transposed V): element e -> k = 16g + e.
// Caller passes the per-lane base (already includes column*ld + 16g).
__device__ inline v16bf load_frag_contig(const unsigned short* p) {
    FragU f;
    f.q[0] = *(const uint4*)(p);
    f.q[1] = *(const uint4*)(p + 8);
    return f.v;
}

// --- kernel 1a: fp32 -> bf16 (no transpose, for x) ------------------------

__global__ void cvt_bf16_kernel(const float* __restrict__ in,
                                unsigned short* __restrict__ out, int n) {
    int i = blockIdx.x * blockDim.x + threadIdx.x;
    int stride = gridDim.x * blockDim.x;
    for (; i < n; i += stride) out[i] = f2bf(in[i]);
}

// --- kernel 1b: fp32 [batch, R, C] -> bf16 [batch, C, R] (weights) --------

__global__ void cvt_bf16_transpose_kernel(const float* __restrict__ in,
                                          unsigned short* __restrict__ out,
                                          int R, int C, int total) {
    int i = blockIdx.x * blockDim.x + threadIdx.x;
    int stride = gridDim.x * blockDim.x;
    for (; i < total; i += stride) {
        int b = i / (R * C);
        int r = (i / C) % R;
        int c = i % C;
        out[((size_t)b * C + c) * R + r] = f2bf(in[i]);
    }
}

// --- kernel 2: per-head projection  out[b,h] = (x[b] @ W[h] + bias)*scale -
// GEMM M=S=2048, K=D=1024, N=HD=64. 4 waves/WG, wave -> 32x64 tile.
// Weights pre-transposed: Wt [H, HD, D]  (k contiguous).
// TOUT=false: out [B*H, S, HD] ; TOUT=true (for V): out [B*H, HD, S].

template <bool TOUT>
__global__ __launch_bounds__(128) void qkv_gemm_kernel(
    const unsigned short* __restrict__ xb,   // [B,S,D] bf16
    const unsigned short* __restrict__ Wt,   // [H,HD,D] bf16
    const float*          __restrict__ bias, // [H,HD] f32
    unsigned short*       __restrict__ outb,
    float scale)
{
    const int lane = threadIdx.x & 31;
    const int wave = threadIdx.x >> 5;
    const int bh = blockIdx.y;
    const int b = bh / HH, h = bh % HH;
    const int m0 = blockIdx.x * 128 + wave * 32;
    const int n = lane & 15, g = lane >> 4;

    const unsigned short* Abase = xb + ((size_t)b * SS + m0) * DDIM;
    const unsigned short* Wh    = Wt + (size_t)h * HDIM * DDIM;

    v8f acc[2][4] = {};
    for (int k0 = 0; k0 < DDIM; k0 += 32) {
        __builtin_prefetch(Abase + (size_t)(lane & 15) * DDIM + k0 + 128, 0, 0);
        v16bf a0 = load_frag_A(Abase,              DDIM, k0, lane);
        v16bf a1 = load_frag_A(Abase + 16 * DDIM,  DDIM, k0, lane);
#pragma unroll
        for (int nf = 0; nf < 4; ++nf) {
            v16bf bf = load_frag_contig(Wh + (size_t)(nf * 16 + n) * DDIM + k0 + 16 * g);
            acc[0][nf] = __builtin_amdgcn_wmma_f32_16x16x32_bf16(
                false, a0, false, bf, (short)0, acc[0][nf], false, false);
            acc[1][nf] = __builtin_amdgcn_wmma_f32_16x16x32_bf16(
                false, a1, false, bf, (short)0, acc[1][nf], false, false);
        }
    }

#pragma unroll
    for (int mt = 0; mt < 2; ++mt) {
#pragma unroll
        for (int nf = 0; nf < 4; ++nf) {
            float bv = bias[h * HDIM + nf * 16 + n];
#pragma unroll
            for (int r = 0; r < 8; ++r) {
                float v = (acc[mt][nf][r] + bv) * scale;
                int s = m0 + mt * 16 + r + 8 * g;
                if (TOUT) {  // transposed: [B*H, HD, S]
                    outb[((size_t)bh * HDIM + nf * 16 + n) * SS + s] = f2bf(v);
                } else {     // row-major: [B*H, S, HD]
                    outb[((size_t)bh * SS + s) * HDIM + nf * 16 + n] = f2bf(v);
                }
            }
        }
    }
}

// --- kernel 3: flash attention, one wave per 16-query tile, 64 keys/iter --

__global__ __launch_bounds__(128) void attn_kernel(
    const unsigned short* __restrict__ Qb,  // [B*H,S,HD] bf16 (pre-scaled)
    const unsigned short* __restrict__ Kb,  // [B*H,S,HD] bf16
    const unsigned short* __restrict__ Vt,  // [B*H,HD,S] bf16 (transposed)
    unsigned short*       __restrict__ Ob)  // [B,S,H*HD] bf16 (head concat)
{
    __shared__ unsigned short plds[4 * 16 * 64];  // bf16 P tiles, 2KB/wave
    const int lane = threadIdx.x & 31;
    const int wave = threadIdx.x >> 5;
    const int bh = blockIdx.y;
    const int b = bh / HH, h = bh % HH;
    const int q0 = (blockIdx.x * 4 + wave) * 16;
    const int n = lane & 15, g = lane >> 4;
    unsigned short* P = plds + wave * 16 * 64;

    const unsigned short* Qbase = Qb + ((size_t)bh * SS + q0) * HDIM;
    const unsigned short* Kbase = Kb + (size_t)bh * SS * HDIM;
    const unsigned short* Vbase = Vt + (size_t)bh * HDIM * SS;

    v16bf aq0 = load_frag_A(Qbase, HDIM, 0,  lane);   // head dims 0..31
    v16bf aq1 = load_frag_A(Qbase, HDIM, 32, lane);   // head dims 32..63

    v8f o[4] = {};
    float Mrow[8], Lrow[8];
#pragma unroll
    for (int r = 0; r < 8; ++r) { Mrow[r] = -3.0e38f; Lrow[r] = 0.f; }

    for (int kt = 0; kt < SS; kt += 64) {
        // scores for 64 keys = four 16x16 C tiles, 2 WMMAs each (k=32+32)
        v8f s[4];
#pragma unroll
        for (int t = 0; t < 4; ++t) {
            const unsigned short* kr = Kbase + (size_t)(kt + t * 16 + n) * HDIM + 16 * g;
            v16bf b0 = load_frag_contig(kr);
            v16bf b1 = load_frag_contig(kr + 32);
            v8f z = {};
            z = __builtin_amdgcn_wmma_f32_16x16x32_bf16(false, aq0, false, b0, (short)0, z, false, false);
            z = __builtin_amdgcn_wmma_f32_16x16x32_bf16(false, aq1, false, b1, (short)0, z, false, false);
            s[t] = z;
        }

        // online softmax update; row r -> C-vgpr r (row r+8g), cols across
        // the 16-lane half -> one shfl_xor butterfly per 64 keys per row
#pragma unroll
        for (int r = 0; r < 8; ++r) {
            float mx = fmaxf(fmaxf(s[0][r], s[1][r]), fmaxf(s[2][r], s[3][r]));
            mx = fmaxf(mx, __shfl_xor(mx, 1));
            mx = fmaxf(mx, __shfl_xor(mx, 2));
            mx = fmaxf(mx, __shfl_xor(mx, 4));
            mx = fmaxf(mx, __shfl_xor(mx, 8));
            float newM = fmaxf(Mrow[r], mx);
            float corr = __expf(Mrow[r] - newM);
            float p0 = __expf(s[0][r] - newM);
            float p1 = __expf(s[1][r] - newM);
            float p2 = __expf(s[2][r] - newM);
            float p3 = __expf(s[3][r] - newM);
            float rs = (p0 + p1) + (p2 + p3);
            rs += __shfl_xor(rs, 1);
            rs += __shfl_xor(rs, 2);
            rs += __shfl_xor(rs, 4);
            rs += __shfl_xor(rs, 8);
            Lrow[r] = Lrow[r] * corr + rs;
            Mrow[r] = newM;
#pragma unroll
            for (int f = 0; f < 4; ++f) o[f][r] *= corr;
            unsigned short* pw = P + (r + 8 * g) * 64 + n;
            pw[0]  = f2bf(p0);
            pw[16] = f2bf(p1);
            pw[32] = f2bf(p2);
            pw[48] = f2bf(p3);
        }
        asm volatile("s_wait_dscnt 0" ::: "memory");

        // re-read P as two 16x32 bf16 A-fragments: four ds_load_b128 per lane
        v16bf pa0, pa1;
        {
            const unsigned short* pr = P + (lane & 15) * 64;
            FragU f0, f1;
            f0.q[0] = *(const uint4*)(pr + 16 * g);
            f0.q[1] = *(const uint4*)(pr + 8 + 16 * g);
            f1.q[0] = *(const uint4*)(pr + 32 + 16 * g);
            f1.q[1] = *(const uint4*)(pr + 40 + 16 * g);
            pa0 = f0.v;
            pa1 = f1.v;
        }

        // O(16x64) += P(16x64) @ V(64x64); V transposed -> contiguous k
#pragma unroll
        for (int f = 0; f < 4; ++f) {
            const unsigned short* vr = Vbase + (size_t)(f * 16 + n) * SS + kt + 16 * g;
            v16bf vb0 = load_frag_contig(vr);
            v16bf vb1 = load_frag_contig(vr + 32);
            o[f] = __builtin_amdgcn_wmma_f32_16x16x32_bf16(
                false, pa0, false, vb0, (short)0, o[f], false, false);
            o[f] = __builtin_amdgcn_wmma_f32_16x16x32_bf16(
                false, pa1, false, vb1, (short)0, o[f], false, false);
        }
    }

    // normalize and store head-concatenated bf16
    unsigned short* Obase = Ob + ((size_t)b * SS + q0) * DDIM + h * HDIM;
#pragma unroll
    for (int f = 0; f < 4; ++f) {
#pragma unroll
        for (int r = 0; r < 8; ++r) {
            float v = o[f][r] / Lrow[r];
            Obase[(size_t)(r + 8 * g) * DDIM + f * 16 + n] = f2bf(v);
        }
    }
}

// --- kernel 4: output projection  out = O @ Wo + bo  (f32 out) -----------
// Wo pre-transposed: WoT [D_out, D_in] (k contiguous). Wave -> 32x64 tile.

__global__ __launch_bounds__(128) void out_proj_kernel(
    const unsigned short* __restrict__ Ob,   // [B*S, D] bf16
    const unsigned short* __restrict__ WoT,  // [D, D] bf16 transposed
    const float*          __restrict__ bo,   // [D]
    float*                __restrict__ out)  // [B*S, D] f32
{
    const int lane = threadIdx.x & 31;
    const int wave = threadIdx.x >> 5;
    const int m0 = blockIdx.x * 128 + wave * 32;
    const int n0 = blockIdx.y * 64;
    const int n = lane & 15, g = lane >> 4;

    const unsigned short* Abase = Ob + (size_t)m0 * DDIM;

    v8f acc[2][4] = {};
    for (int k0 = 0; k0 < DDIM; k0 += 32) {
        __builtin_prefetch(Abase + (size_t)(lane & 15) * DDIM + k0 + 128, 0, 0);
        v16bf a0 = load_frag_A(Abase,             DDIM, k0, lane);
        v16bf a1 = load_frag_A(Abase + 16 * DDIM, DDIM, k0, lane);
#pragma unroll
        for (int nf = 0; nf < 4; ++nf) {
            v16bf bf = load_frag_contig(WoT + (size_t)(n0 + nf * 16 + n) * DDIM + k0 + 16 * g);
            acc[0][nf] = __builtin_amdgcn_wmma_f32_16x16x32_bf16(
                false, a0, false, bf, (short)0, acc[0][nf], false, false);
            acc[1][nf] = __builtin_amdgcn_wmma_f32_16x16x32_bf16(
                false, a1, false, bf, (short)0, acc[1][nf], false, false);
        }
    }

#pragma unroll
    for (int mt = 0; mt < 2; ++mt) {
#pragma unroll
        for (int nf = 0; nf < 4; ++nf) {
            float bv = bo[n0 + nf * 16 + n];
#pragma unroll
            for (int r = 0; r < 8; ++r)
                out[(size_t)(m0 + mt * 16 + r + 8 * g) * DDIM + n0 + nf * 16 + n] =
                    acc[mt][nf][r] + bv;
        }
    }
}

// --- launch ---------------------------------------------------------------

extern "C" void kernel_launch(void* const* d_in, const int* in_sizes, int n_in,
                              void* d_out, int out_size, void* d_ws, size_t ws_size,
                              hipStream_t stream) {
    const float* x  = (const float*)d_in[0];
    const float* Wq = (const float*)d_in[1];
    const float* bq = (const float*)d_in[2];
    const float* Wk = (const float*)d_in[3];
    const float* bk = (const float*)d_in[4];
    const float* Wv = (const float*)d_in[5];
    const float* bv = (const float*)d_in[6];
    const float* Wo = (const float*)d_in[7];
    const float* bo = (const float*)d_in[8];
    float* out = (float*)d_out;

    char* ws = (char*)d_ws;
    size_t off = 0;
    auto alloc_us = [&](size_t n_elems) -> unsigned short* {
        unsigned short* p = (unsigned short*)(ws + off);
        off += n_elems * sizeof(unsigned short);
        return p;
    };
    const size_t nX = (size_t)BB * SS * DDIM;       // 4M elems
    const size_t nW = (size_t)HH * DDIM * HDIM;     // 1M elems
    const size_t nQ = (size_t)BB * HH * SS * HDIM;  // 4M elems
    unsigned short* xb  = alloc_us(nX);
    unsigned short* Wqt = alloc_us(nW);             // [H,HD,D]
    unsigned short* Wkt = alloc_us(nW);
    unsigned short* Wvt = alloc_us(nW);
    unsigned short* WoT = alloc_us((size_t)DDIM * DDIM);
    unsigned short* Qb  = alloc_us(nQ);             // [B*H,S,HD]
    unsigned short* Kb  = alloc_us(nQ);             // [B*H,S,HD]
    unsigned short* Vtb = alloc_us(nQ);             // [B*H,HD,S]
    unsigned short* Obf = alloc_us(nX);             // [B,S,D]

    // fp32 -> bf16 staging (weights transposed so WMMA B-operands are
    // contiguous along k -> global_load_b128 fragment loads)
    cvt_bf16_kernel<<<2048, 256, 0, stream>>>(x, xb, (int)nX);
    cvt_bf16_transpose_kernel<<<1024, 256, 0, stream>>>(Wq, Wqt, DDIM, HDIM, (int)nW);
    cvt_bf16_transpose_kernel<<<1024, 256, 0, stream>>>(Wk, Wkt, DDIM, HDIM, (int)nW);
    cvt_bf16_transpose_kernel<<<1024, 256, 0, stream>>>(Wv, Wvt, DDIM, HDIM, (int)nW);
    cvt_bf16_transpose_kernel<<<1024, 256, 0, stream>>>(Wo, WoT, DDIM, DDIM, DDIM * DDIM);

    dim3 gQKV(SS / 128, BB * HH);
    qkv_gemm_kernel<false><<<gQKV, 128, 0, stream>>>(xb, Wqt, bq, Qb, SM_SCALE);
    qkv_gemm_kernel<false><<<gQKV, 128, 0, stream>>>(xb, Wkt, bk, Kb, 1.0f);
    qkv_gemm_kernel<true ><<<gQKV, 128, 0, stream>>>(xb, Wvt, bv, Vtb, 1.0f);

    dim3 gAtt(SS / 64, BB * HH);
    attn_kernel<<<gAtt, 128, 0, stream>>>(Qb, Kb, Vtb, Obf);

    dim3 gProj((BB * SS) / 128, DDIM / 64);
    out_proj_kernel<<<gProj, 128, 0, stream>>>(Obf, WoT, bo, out);
}
